// MultiHeadLatentAttention_40630390620925
// MI455X (gfx1250) — matrix operand order repaired and checked by reading, not compiled, not run on hardware
//
#include <hip/hip_runtime.h>

#define BS 16
#define SEQL 512
#define HID 2048
#define DN 512
#define UPD 1024
#define NHEAD 16
#define RHDIM 64
#define VH 64
#define MTOK (BS * SEQL)   // 8192

typedef __attribute__((ext_vector_type(16))) __bf16 v16bf;
typedef __attribute__((ext_vector_type(8)))  float  v8f;

union BFrag {
  v16bf v;
  uint4 q[2];
  unsigned short s[16];
};

__device__ __forceinline__ unsigned short f2bf(float f) {
  union { float f; unsigned int u; } c;
  c.f = f;
  unsigned int u = c.u;
  unsigned int r = (u + 0x7FFFu + ((u >> 16) & 1u)) >> 16;
  return (unsigned short)r;
}
__device__ __forceinline__ float bf2f(unsigned short h) {
  union { float f; unsigned int u; } c;
  c.u = ((unsigned int)h) << 16;
  return c.f;
}

// CDNA5 async global->LDS copy (ASYNCcnt path). ldsAddr = per-lane LDS byte
// address (low 32 bits of generic pointer), gaddr = per-lane 64-bit global.
__device__ __forceinline__ void async_load_b128(unsigned ldsAddr,
                                                unsigned long long gaddr) {
  asm volatile("global_load_async_to_lds_b128 %0, %1, off"
               :
               : "v"(ldsAddr), "v"(gaddr)
               : "memory");
}
__device__ __forceinline__ void wait_async_le1() {
  asm volatile("s_wait_asynccnt 0x1" ::: "memory");
}
__device__ __forceinline__ void wait_async_0() {
  asm volatile("s_wait_asynccnt 0x0" ::: "memory");
}

// ---------------------------------------------------------------- converts
__global__ void cvt_f32_bf16(const float* __restrict__ in,
                             unsigned short* __restrict__ out, long long n) {
  long long i = (long long)blockIdx.x * blockDim.x + threadIdx.x;
  long long stride = (long long)gridDim.x * blockDim.x;
  for (; i < n; i += stride) out[i] = f2bf(in[i]);
}

// in: [K,N] fp32 (row-major, fan_in x fan_out) -> out: [N,K] bf16
__global__ void transpose_f32_bf16(const float* __restrict__ in,
                                   unsigned short* __restrict__ out,
                                   int K, int N) {
  long long total = (long long)K * N;
  long long i = (long long)blockIdx.x * blockDim.x + threadIdx.x;
  long long stride = (long long)gridDim.x * blockDim.x;
  for (; i < total; i += stride) {
    int k = (int)(i % K);
    int n = (int)(i / K);
    out[i] = f2bf(in[(long long)k * N + n]);
  }
}

// ---------------------------------------------------------------- WMMA GEMM
// Y[M,N] = A[M,K](bf16) * Wt[N,K]^T(bf16) + bias ; out fp32 or bf16.
// Block = 8 waves; block tile = 256(M) x 64(N); wave tile = 32(M) x 64(N).
// B tile (64x32 bf16 = 4KB) staged in LDS via async copies, double buffered.
__global__ __launch_bounds__(256) void wmma_gemm_bf16(
    const unsigned short* __restrict__ A,
    const unsigned short* __restrict__ Wt,
    const float* __restrict__ bias,
    void* __restrict__ out, int outBf16,
    int M, int N, int K) {
  __shared__ __align__(16) unsigned short ldsB[2][64 * 32];  // 2 x 4KB

  const int tid  = threadIdx.x;
  const int lane = tid & 31;
  const int wave = tid >> 5;
  const int ml = lane & 15;
  const int g  = lane >> 4;
  const int n0 = blockIdx.x * 64;
  const int m0 = (blockIdx.y * 8 + wave) * 32;

  const long long aRow0 = (long long)(m0 + ml) * K;
  const long long aRow1 = (long long)(m0 + 16 + ml) * K;

  // async-copy chunk owned by this thread: 16B of the 64x32 B tile
  const int nl = tid >> 2;          // 0..63 (row of B tile)
  const int kp = tid & 3;           // 0..3  (16B chunk within 32-elem row)
  const unsigned ldsBase =
      (unsigned)(unsigned long long)(uintptr_t)(&ldsB[0][0]) +
      (unsigned)(tid * 16);
  const unsigned long long gRow =
      (unsigned long long)(uintptr_t)(Wt + (long long)(n0 + nl) * K + kp * 8);

  v8f acc0[4] = {};
  v8f acc1[4] = {};

  // prologue: kick off the first B tile
  async_load_b128(ldsBase, gRow);

  for (int kb = 0; kb < K; kb += 32) {
    const int cur = (kb >> 5) & 1;
    const bool hasNext = (kb + 32) < K;
    if (hasNext)
      async_load_b128(ldsBase + (unsigned)((cur ^ 1) * 4096),
                      gRow + 2ull * (kb + 32));

    // A fragments for this k-step (global; overlaps the async wait below)
    BFrag a0, a1;
    a0.q[0] = *(const uint4*)(A + aRow0 + kb + 8 * g);
    a0.q[1] = *(const uint4*)(A + aRow0 + kb + 16 + 8 * g);
    a1.q[0] = *(const uint4*)(A + aRow1 + kb + 8 * g);
    a1.q[1] = *(const uint4*)(A + aRow1 + kb + 16 + 8 * g);

    if (hasNext) wait_async_le1(); else wait_async_0();
    __syncthreads();  // B tile [cur] visible to all waves

    // B fragments from LDS: row = t*16+ml, 16B at k=8*g and 16B at k=16+8*g
    BFrag b[4];
#pragma unroll
    for (int t = 0; t < 4; ++t) {
      const unsigned short* bp = &ldsB[cur][(t * 16 + ml) * 32 + 8 * g];
      b[t].q[0] = ((const uint4*)bp)[0];
      b[t].q[1] = *(const uint4*)(bp + 16);
    }
#pragma unroll
    for (int t = 0; t < 4; ++t) {
      acc0[t] = __builtin_amdgcn_wmma_f32_16x16x32_bf16(
          false, a0.v, false, b[t].v, (short)0, acc0[t], false, false);
      acc1[t] = __builtin_amdgcn_wmma_f32_16x16x32_bf16(
          false, a1.v, false, b[t].v, (short)0, acc1[t], false, false);
    }
    __syncthreads();  // all waves done reading [cur] before it is re-filled
  }

#pragma unroll
  for (int t = 0; t < 4; ++t) {
    const int col = n0 + t * 16 + ml;
    const float bv = bias[col];
#pragma unroll
    for (int i = 0; i < 8; ++i) {
      const int row0 = m0 + i + 8 * g;
      const int row1 = m0 + 16 + i + 8 * g;
      const float v0 = acc0[t][i] + bv;
      const float v1 = acc1[t][i] + bv;
      if (outBf16) {
        ((unsigned short*)out)[(long long)row0 * N + col] = f2bf(v0);
        ((unsigned short*)out)[(long long)row1 * N + col] = f2bf(v1);
      } else {
        ((float*)out)[(long long)row0 * N + col] = v0;
        ((float*)out)[(long long)row1 * N + col] = v1;
      }
    }
  }
}

// ---------------------------------------------------------------- assemble
// qc/qr/kc/vc: [MTOK, 1024] bf16 (col = h*64 + d); kr: [MTOK, 64] bf16
// qcat/kcat: [BS, NHEAD, SEQL, 128] bf16 ; vT: [BS, NHEAD, 64, SEQL] bf16
__global__ void assemble_kernel(const unsigned short* __restrict__ qc,
                                const unsigned short* __restrict__ qr,
                                const unsigned short* __restrict__ kc,
                                const unsigned short* __restrict__ kr,
                                const unsigned short* __restrict__ vc,
                                unsigned short* __restrict__ qcat,
                                unsigned short* __restrict__ kcat,
                                unsigned short* __restrict__ vT) {
  long long idx = (long long)blockIdx.x * blockDim.x + threadIdx.x;
  const int p = (int)(idx & 31);           // pair index 0..31
  long long t = idx >> 5;
  const int s = (int)(t & (SEQL - 1));
  t >>= 9;
  const int h = (int)(t & (NHEAD - 1));
  const int b = (int)(t >> 4);
  if (b >= BS) return;

  const long long tok = (long long)b * SEQL + s;
  const long long inbase = tok * (NHEAD * 64) + h * 64;
  const long long obase = (((long long)(b * NHEAD + h)) * SEQL + s) * 128;

  // content halves
  qcat[obase + 2 * p]     = qc[inbase + 2 * p];
  qcat[obase + 2 * p + 1] = qc[inbase + 2 * p + 1];
  kcat[obase + 2 * p]     = kc[inbase + 2 * p];
  kcat[obase + 2 * p + 1] = kc[inbase + 2 * p + 1];

  // rope (reference quirk: cos_pos = sin(pos), sin_pos = cos(pos))
  const float LOG1E4_OVER32 = 9.210340371976184f / 32.0f;
  const float theta = __expf(-(float)p * LOG1E4_OVER32);
  const float ang = (float)s * theta;
  const float sn = __sinf(ang);
  const float cs = __cosf(ang);

  {
    float q0 = bf2f(qr[inbase + 2 * p]);
    float q1 = bf2f(qr[inbase + 2 * p + 1]);
    qcat[obase + 64 + 2 * p]     = f2bf(q0 * sn - q1 * cs);
    qcat[obase + 64 + 2 * p + 1] = f2bf(q1 * sn + q0 * cs);
  }
  {
    float k0 = bf2f(kr[tok * RHDIM + 2 * p]);
    float k1 = bf2f(kr[tok * RHDIM + 2 * p + 1]);
    kcat[obase + 64 + 2 * p]     = f2bf(k0 * sn - k1 * cs);
    kcat[obase + 64 + 2 * p + 1] = f2bf(k1 * sn + k0 * cs);
  }
  // v transpose: vT[b,h,d,s]
  const long long vb = ((long long)(b * NHEAD + h)) * VH * SEQL;
  vT[vb + (long long)(2 * p) * SEQL + s]     = vc[inbase + 2 * p];
  vT[vb + (long long)(2 * p + 1) * SEQL + s] = vc[inbase + 2 * p + 1];
}

// ---------------------------------------------------------------- attention
// one wave per (b, h, q-tile of 16). qcat/kcat [b,h,s,128]; vT [b,h,64,s].
// ctx out: [BS*SEQL, NHEAD*VH] bf16 (token-major for the final GEMM)
__global__ __launch_bounds__(32) void mla_attn(
    const unsigned short* __restrict__ qcat,
    const unsigned short* __restrict__ kcat,
    const unsigned short* __restrict__ vT,
    const int* __restrict__ mask,
    unsigned short* __restrict__ ctx) {
  __shared__ __align__(16) float sc[16][SEQL];          // 32 KB
  __shared__ __align__(16) unsigned short P[16][SEQL];  // 16 KB

  const int lane = threadIdx.x;
  const int ml = lane & 15;
  const int g = lane >> 4;
  int task = blockIdx.x;
  const int qt = task & 31;  task >>= 5;
  const int h = task & 15;
  const int b = task >> 4;

  const long long bh = (long long)(b * NHEAD + h);
  const unsigned short* qb = qcat + (bh * SEQL + qt * 16 + ml) * 128;
  BFrag aq[4];
#pragma unroll
  for (int kc = 0; kc < 4; ++kc) {
    aq[kc].q[0] = *(const uint4*)(qb + kc * 32 + 8 * g);
    aq[kc].q[1] = *(const uint4*)(qb + kc * 32 + 16 + 8 * g);
  }

  const unsigned short* kb = kcat + bh * SEQL * 128;
  const float SCALE = 0.051776695296636886f;  // 1/(sqrt(128)+sqrt(64))
  for (int kt = 0; kt < SEQL / 16; ++kt) {
    const unsigned short* kp = kb + (long long)(kt * 16 + ml) * 128 + 8 * g;
    v8f c = {};
#pragma unroll
    for (int kc = 0; kc < 4; ++kc) {
      BFrag bk;
      bk.q[0] = ((const uint4*)(kp + kc * 32))[0];
      bk.q[1] = *(const uint4*)(kp + kc * 32 + 16);
      c = __builtin_amdgcn_wmma_f32_16x16x32_bf16(
          false, aq[kc].v, false, bk.v, (short)0, c, false, false);
    }
#pragma unroll
    for (int i = 0; i < 8; ++i)
      sc[i + 8 * g][kt * 16 + ml] = c[i] * SCALE;
  }
  __syncthreads();

  // softmax: lane owns (row = ml, cols [g*256, g*256+256))
  const int r = ml;
  const int c0 = g * 256;
  const int* mrow = mask + b * SEQL;
  float mx = -__builtin_inff();
  for (int j = 0; j < 256; ++j) {
    const int col = c0 + j;
    float s = sc[r][col];
    if (mrow[col] == 0) { s = -__builtin_inff(); sc[r][col] = s; }
    mx = fmaxf(mx, s);
  }
  mx = fmaxf(mx, __shfl_xor(mx, 16, 32));
  float sum = 0.0f;
  for (int j = 0; j < 256; ++j) {
    const int col = c0 + j;
    const float p = __expf(sc[r][col] - mx);
    sc[r][col] = p;
    sum += p;
  }
  sum += __shfl_xor(sum, 16, 32);
  const float inv = 1.0f / sum;
  for (int j = 0; j < 256; ++j) {
    const int col = c0 + j;
    P[r][col] = f2bf(sc[r][col] * inv);
  }
  __syncthreads();

  // ctx = P[16x512] @ V[512x64]  (B-frags from vT[64,512])
  const unsigned short* vb = vT + bh * VH * SEQL;
  v8f acc[4] = {};
  for (int ks = 0; ks < SEQL / 32; ++ks) {
    BFrag a;
    const unsigned short* pp = &P[ml][ks * 32 + 8 * g];
    a.q[0] = ((const uint4*)pp)[0];
    a.q[1] = *(const uint4*)(pp + 16);
#pragma unroll
    for (int t = 0; t < 4; ++t) {
      const unsigned short* vp =
          vb + (long long)(t * 16 + ml) * SEQL + ks * 32 + 8 * g;
      BFrag bv;
      bv.q[0] = ((const uint4*)vp)[0];
      bv.q[1] = *(const uint4*)(vp + 16);
      acc[t] = __builtin_amdgcn_wmma_f32_16x16x32_bf16(
          false, a.v, false, bv.v, (short)0, acc[t], false, false);
    }
  }
#pragma unroll
  for (int t = 0; t < 4; ++t) {
#pragma unroll
    for (int i = 0; i < 8; ++i) {
      const int q = qt * 16 + i + 8 * g;
      ctx[((long long)(b * SEQL + q)) * (NHEAD * VH) + h * VH + t * 16 + ml] =
          f2bf(acc[t][i]);
    }
  }
}

// ---------------------------------------------------------------- launch
extern "C" void kernel_launch(void* const* d_in, const int* in_sizes, int n_in,
                              void* d_out, int out_size, void* d_ws,
                              size_t ws_size, hipStream_t stream) {
  (void)in_sizes; (void)n_in; (void)out_size; (void)ws_size;
  const float* x     = (const float*)d_in[0];
  const int*   mask  = (const int*)d_in[1];
  const float* w_dkv = (const float*)d_in[2];  const float* b_dkv = (const float*)d_in[3];
  const float* w_dq  = (const float*)d_in[4];  const float* b_dq  = (const float*)d_in[5];
  const float* w_uq  = (const float*)d_in[6];  const float* b_uq  = (const float*)d_in[7];
  const float* w_uk  = (const float*)d_in[8];  const float* b_uk  = (const float*)d_in[9];
  const float* w_uv  = (const float*)d_in[10]; const float* b_uv  = (const float*)d_in[11];
  const float* w_qr  = (const float*)d_in[12]; const float* b_qr  = (const float*)d_in[13];
  const float* w_kr  = (const float*)d_in[14]; const float* b_kr  = (const float*)d_in[15];
  const float* w_fc  = (const float*)d_in[16]; const float* b_fc  = (const float*)d_in[17];
  float* out = (float*)d_out;

  char* ws = (char*)d_ws;
  size_t off = 0;
  auto take = [&](size_t bytes) -> char* {
    char* p = ws + off;
    off = (off + bytes + 255) & ~(size_t)255;
    return p;
  };

  unsigned short* xb    = (unsigned short*)take((size_t)MTOK * HID * 2);
  unsigned short* wdkvT = (unsigned short*)take((size_t)DN * HID * 2);
  unsigned short* wdqT  = (unsigned short*)take((size_t)DN * HID * 2);
  unsigned short* wuqT  = (unsigned short*)take((size_t)UPD * DN * 2);
  unsigned short* wukT  = (unsigned short*)take((size_t)UPD * DN * 2);
  unsigned short* wuvT  = (unsigned short*)take((size_t)UPD * DN * 2);
  unsigned short* wqrT  = (unsigned short*)take((size_t)UPD * DN * 2);
  unsigned short* wkrT  = (unsigned short*)take((size_t)RHDIM * DN * 2);
  unsigned short* wfcT  = (unsigned short*)take((size_t)HID * UPD * 2);
  unsigned short* ckv   = (unsigned short*)take((size_t)MTOK * DN * 2);
  unsigned short* cq    = (unsigned short*)take((size_t)MTOK * DN * 2);
  unsigned short* qcb   = (unsigned short*)take((size_t)MTOK * UPD * 2);
  unsigned short* kcb   = (unsigned short*)take((size_t)MTOK * UPD * 2);
  unsigned short* vcb   = (unsigned short*)take((size_t)MTOK * UPD * 2);
  unsigned short* qrb   = (unsigned short*)take((size_t)MTOK * UPD * 2);
  unsigned short* krb   = (unsigned short*)take((size_t)MTOK * RHDIM * 2);
  unsigned short* kcat  = (unsigned short*)take((size_t)BS * NHEAD * SEQL * 128 * 2);
  unsigned short* vT    = (unsigned short*)take((size_t)BS * NHEAD * VH * SEQL * 2);
  unsigned short* qcat  = xb;   // alias: xb dead after the two down-proj GEMMs
  unsigned short* ctx   = ckv;  // alias: spans ckv+cq, dead after k_r GEMM

  // 1) convert x, transpose weights to [N,K] bf16
  cvt_f32_bf16<<<4096, 256, 0, stream>>>(x, xb, (long long)MTOK * HID);
  transpose_f32_bf16<<<2048, 256, 0, stream>>>(w_dkv, wdkvT, HID, DN);
  transpose_f32_bf16<<<2048, 256, 0, stream>>>(w_dq,  wdqT,  HID, DN);
  transpose_f32_bf16<<<1024, 256, 0, stream>>>(w_uq,  wuqT,  DN, UPD);
  transpose_f32_bf16<<<1024, 256, 0, stream>>>(w_uk,  wukT,  DN, UPD);
  transpose_f32_bf16<<<1024, 256, 0, stream>>>(w_uv,  wuvT,  DN, UPD);
  transpose_f32_bf16<<<1024, 256, 0, stream>>>(w_qr,  wqrT,  DN, UPD);
  transpose_f32_bf16<<<128,  256, 0, stream>>>(w_kr,  wkrT,  DN, RHDIM);
  transpose_f32_bf16<<<4096, 256, 0, stream>>>(w_fc,  wfcT,  UPD, HID);

  // 2) latent down-projections (block tile 256x64)
  dim3 gDown(DN / 64, MTOK / 256);
  wmma_gemm_bf16<<<gDown, 256, 0, stream>>>(xb, wdkvT, b_dkv, ckv, 1, MTOK, DN, HID);
  wmma_gemm_bf16<<<gDown, 256, 0, stream>>>(xb, wdqT,  b_dq,  cq,  1, MTOK, DN, HID);

  // 3) up-projections
  dim3 gUp(UPD / 64, MTOK / 256);
  wmma_gemm_bf16<<<gUp, 256, 0, stream>>>(ckv, wukT, b_uk, kcb, 1, MTOK, UPD, DN);
  wmma_gemm_bf16<<<gUp, 256, 0, stream>>>(ckv, wuvT, b_uv, vcb, 1, MTOK, UPD, DN);
  wmma_gemm_bf16<<<gUp, 256, 0, stream>>>(cq,  wuqT, b_uq, qcb, 1, MTOK, UPD, DN);
  wmma_gemm_bf16<<<gUp, 256, 0, stream>>>(cq,  wqrT, b_qr, qrb, 1, MTOK, UPD, DN);
  dim3 gKr(RHDIM / 64, MTOK / 256);
  wmma_gemm_bf16<<<gKr, 256, 0, stream>>>(ckv, wkrT, b_kr, krb, 1, MTOK, RHDIM, DN);

  // 4) rope + concat + v-transpose
  assemble_kernel<<<(BS * NHEAD * SEQL * 32) / 256, 256, 0, stream>>>(
      qcb, qrb, kcb, krb, vcb, qcat, kcat, vT);

  // 5) fused attention (softmax in LDS, both GEMMs on WMMA)
  mla_attn<<<BS * NHEAD * (SEQL / 16), 32, 0, stream>>>(qcat, kcat, vT, mask, ctx);

  // 6) output projection -> fp32 d_out
  dim3 gFc(HID / 64, MTOK / 256);
  wmma_gemm_bf16<<<gFc, 256, 0, stream>>>(ctx, wfcT, b_fc, out, 0, MTOK, HID, UPD);
}
